// GATCooccurrenceModel_25013889532275
// MI455X (gfx1250) — matrix operand order repaired and checked by reading, not compiled, not run on hardware
//
#include <hip/hip_runtime.h>
#include <hip/hip_bf16.h>

// ---------------------------------------------------------------------------
// GAT co-occurrence model for MI455X (gfx1250, wave32, WMMA).
// Heavy lifting = 3 GEMMs on v_wmma_f32_16x16x32_bf16 with double-buffered
// LDS staging (async global->LDS copies when the toolchain exposes them).
// ---------------------------------------------------------------------------

typedef __attribute__((ext_vector_type(16))) __bf16 v16bf;
typedef __attribute__((ext_vector_type(8)))  float  v8f;
typedef __attribute__((ext_vector_type(4)))  int    v4i;

union FragBF { v16bf v; unsigned int u32[8]; };

#define BATCH 512
#define CNODES 15
#define NNODES (BATCH * CNODES)

#define TM 128
#define TN 128
#define TK 32

// ---- async global->LDS copy support (guarded; falls back to reg staging) ---
#if defined(__HIP_DEVICE_COMPILE__) && \
    __has_builtin(__builtin_amdgcn_global_load_async_to_lds_b128)
#define USE_ASYNC_LDS 1
#else
#define USE_ASYNC_LDS 0
#endif

#if USE_ASYNC_LDS
__device__ __forceinline__ void async_copy_b128(const void* g, void* l) {
    __builtin_amdgcn_global_load_async_to_lds_b128(
        (__attribute__((address_space(1))) v4i*)(unsigned long long)(uintptr_t)g,
        (__attribute__((address_space(3))) v4i*)(unsigned int)(uintptr_t)l,
        0, 0);
}
__device__ __forceinline__ void async_wait0() {
#if __has_builtin(__builtin_amdgcn_s_wait_asynccnt)
    __builtin_amdgcn_s_wait_asynccnt(0);
#else
    asm volatile("s_wait_asynccnt 0" ::: "memory");
#endif
}
#endif

__device__ __forceinline__ float lrelu02(float x) { return x > 0.f ? x : 0.2f * x; }
__device__ __forceinline__ float elu1(float x)    { return x > 0.f ? x : (__expf(x) - 1.f); }

__device__ __forceinline__ unsigned short f32_to_bf16_rne(float f) {
    unsigned int u = __float_as_uint(f);
    u += 0x7fffu + ((u >> 16) & 1u);
    return (unsigned short)(u >> 16);
}

// ---------------------------------------------------------------------------
// f32 -> bf16 (RNE), grid-stride
// ---------------------------------------------------------------------------
__global__ void cvt_bf16_kernel(const float* __restrict__ in,
                                unsigned short* __restrict__ out, size_t n) {
    size_t i = (size_t)blockIdx.x * blockDim.x + threadIdx.x;
    size_t st = (size_t)gridDim.x * blockDim.x;
    for (; i < n; i += st) out[i] = f32_to_bf16_rne(in[i]);
}

// ---------------------------------------------------------------------------
// Transpose + convert: in f32 (K x N row-major) -> out bf16 (N x K row-major).
// 32x32 LDS tile, 256 threads. K % 32 == 0, N % 32 == 0 for all calls.
// ---------------------------------------------------------------------------
__global__ __launch_bounds__(256)
void cvt_transpose_bf16_kernel(const float* __restrict__ in,
                               unsigned short* __restrict__ out, int K, int N) {
    __shared__ float tile[32][33];
    const int kb = blockIdx.y * 32, nb = blockIdx.x * 32;
    const int tx = threadIdx.x & 31, ty = threadIdx.x >> 5; // ty: 0..7
    for (int i = ty; i < 32; i += 8)
        tile[i][tx] = in[(size_t)(kb + i) * N + nb + tx];
    __syncthreads();
    for (int i = ty; i < 32; i += 8)
        out[(size_t)(nb + i) * K + kb + tx] = f32_to_bf16_rne(tile[tx][i]);
}

// ---------------------------------------------------------------------------
// Tiled bf16 WMMA GEMM:
//   C(MxN,f32) = A(MxK,bf16,rowmaj) @ Bt(NxK,bf16,rowmaj)^T
// Block tile 128x128, K-step 32, double-buffered LDS (32 KB).
// 8 waves; each wave computes a 32x64 strip as 2x4 wmma 16x16 tiles.
// M % 128 == N % 128 == 0, K % 32 == 0 for every call below.
// ---------------------------------------------------------------------------
__global__ __launch_bounds__(256)
void gemm_bf16_wmma_kernel(const unsigned short* __restrict__ A,
                           const unsigned short* __restrict__ Bt,
                           float* __restrict__ C, int M, int N, int K) {
    __shared__ __align__(16) unsigned short As[2][TM][TK]; // 2 x 8 KB
    __shared__ __align__(16) unsigned short Bs[2][TN][TK]; // 2 x 8 KB

    const int tid  = threadIdx.x;
    const int lane = tid & 31;
    const int wave = tid >> 5;
    const int wm   = wave & 3;  // 4 wave rows -> 32 M each
    const int wn   = wave >> 2; // 2 wave cols -> 64 N each
    const int mb   = blockIdx.y * TM;
    const int nb   = blockIdx.x * TN;
    const int l16  = lane & 15;
    const int hsel = lane >> 4;

    const int ntiles = K / TK;

    v8f zero = {0.f, 0.f, 0.f, 0.f, 0.f, 0.f, 0.f, 0.f};
    v8f acc[2][4] = {{zero, zero, zero, zero}, {zero, zero, zero, zero}};

    // per-thread staging chunks: 2 x b128 for A, 2 x b128 for B
    // A chunk p: row = p>>1, 16-short half = p&1
    // B chunk p: row n = p>>2, 8-short quarter = p&3
#if !USE_ASYNC_LDS
    uint4 avreg[2], bvreg[2];
#endif

    // ---- prologue: stage tile 0 into buffer 0 ----
    {
#if USE_ASYNC_LDS
        #pragma unroll
        for (int i = 0; i < 2; ++i) {
            int p = tid + i * 256;
            int r = p >> 1, hf = (p & 1) << 4;
            async_copy_b128(A + (size_t)(mb + r) * K + hf, &As[0][r][hf]);
            int n = p >> 2, q = (p & 3) << 3;
            async_copy_b128(Bt + (size_t)(nb + n) * K + q, &Bs[0][n][q]);
        }
        async_wait0();
#else
        #pragma unroll
        for (int i = 0; i < 2; ++i) {
            int p = tid + i * 256;
            int r = p >> 1, hf = (p & 1) << 4;
            avreg[i] = *reinterpret_cast<const uint4*>(A + (size_t)(mb + r) * K + hf);
            int n = p >> 2, q = (p & 3) << 3;
            bvreg[i] = *reinterpret_cast<const uint4*>(Bt + (size_t)(nb + n) * K + q);
        }
        #pragma unroll
        for (int i = 0; i < 2; ++i) {
            int p = tid + i * 256;
            int r = p >> 1, hf = (p & 1) << 4;
            *reinterpret_cast<uint4*>(&As[0][r][hf]) = avreg[i];
            int n = p >> 2, q = (p & 3) << 3;
            *reinterpret_cast<uint4*>(&Bs[0][n][q]) = bvreg[i];
        }
#endif
    }
    __syncthreads();

    for (int kt = 0; kt < ntiles; ++kt) {
        const int buf = kt & 1;
        const int k0n = (kt + 1) * TK;
        const bool more = (kt + 1) < ntiles;

        // ---- kick off staging of the next tile into the other buffer ----
        if (more) {
#if USE_ASYNC_LDS
            #pragma unroll
            for (int i = 0; i < 2; ++i) {
                int p = tid + i * 256;
                int r = p >> 1, hf = (p & 1) << 4;
                async_copy_b128(A + (size_t)(mb + r) * K + k0n + hf,
                                &As[buf ^ 1][r][hf]);
                int n = p >> 2, q = (p & 3) << 3;
                async_copy_b128(Bt + (size_t)(nb + n) * K + k0n + q,
                                &Bs[buf ^ 1][n][q]);
            }
#else
            #pragma unroll
            for (int i = 0; i < 2; ++i) {
                int p = tid + i * 256;
                int r = p >> 1, hf = (p & 1) << 4;
                avreg[i] = *reinterpret_cast<const uint4*>(
                    A + (size_t)(mb + r) * K + k0n + hf);
                int n = p >> 2, q = (p & 3) << 3;
                bvreg[i] = *reinterpret_cast<const uint4*>(
                    Bt + (size_t)(nb + n) * K + k0n + q);
            }
#endif
        }

        // ---- compute current tile from LDS ----
        FragBF af[2], bfr[4];
        #pragma unroll
        for (int mi = 0; mi < 2; ++mi) {
            int row = wm * 32 + mi * 16 + l16;
            #pragma unroll
            for (int i = 0; i < 8; ++i) {
                int kk = (i < 4) ? (hsel * 8 + 2 * i)
                                 : (16 + hsel * 8 + 2 * (i - 4));
                af[mi].u32[i] =
                    *reinterpret_cast<const unsigned int*>(&As[buf][row][kk]);
            }
        }
        #pragma unroll
        for (int ni = 0; ni < 4; ++ni) {
            int col = wn * 64 + ni * 16 + l16;
            #pragma unroll
            for (int i = 0; i < 8; ++i) {
                int kk = (i < 4) ? (hsel * 8 + 2 * i)
                                 : (16 + hsel * 8 + 2 * (i - 4));
                bfr[ni].u32[i] =
                    *reinterpret_cast<const unsigned int*>(&Bs[buf][col][kk]);
            }
        }
        #pragma unroll
        for (int mi = 0; mi < 2; ++mi)
            #pragma unroll
            for (int ni = 0; ni < 4; ++ni)
                acc[mi][ni] = __builtin_amdgcn_wmma_f32_16x16x32_bf16(
                    false, af[mi].v, false, bfr[ni].v,
                    (short)0, acc[mi][ni], false, false);

        // ---- finish staging next tile ----
        if (more) {
#if USE_ASYNC_LDS
            async_wait0();
#else
            #pragma unroll
            for (int i = 0; i < 2; ++i) {
                int p = tid + i * 256;
                int r = p >> 1, hf = (p & 1) << 4;
                *reinterpret_cast<uint4*>(&As[buf ^ 1][r][hf]) = avreg[i];
                int n = p >> 2, q = (p & 3) << 3;
                *reinterpret_cast<uint4*>(&Bs[buf ^ 1][n][q]) = bvreg[i];
            }
#endif
        }
        __syncthreads();
    }

    // ---- store D (f32 16x16 C layout: vgpr j -> M=j+8*hsel, lane%16 -> N) ----
    #pragma unroll
    for (int mi = 0; mi < 2; ++mi)
        #pragma unroll
        for (int ni = 0; ni < 4; ++ni) {
            int rbase = mb + wm * 32 + mi * 16 + hsel * 8;
            int col   = nb + wn * 64 + ni * 16 + l16;
            #pragma unroll
            for (int j = 0; j < 8; ++j)
                C[(size_t)(rbase + j) * N + col] = acc[mi][ni][j];
        }
}

// ---------------------------------------------------------------------------
// h1[n][j] = Hemb[b][j] + W1[(1024 + c)][j]   (exploits broadcast structure)
// ---------------------------------------------------------------------------
__global__ void build_h1_kernel(const float* __restrict__ Hemb,
                                const float* __restrict__ W1,
                                float* __restrict__ h1, size_t total) {
    size_t i = (size_t)blockIdx.x * blockDim.x + threadIdx.x;
    size_t st = (size_t)gridDim.x * blockDim.x;
    for (; i < total; i += st) {
        int n = (int)(i >> 12);         // / 4096
        int j = (int)(i & 4095);
        int b = n / CNODES;
        int c = n - b * CNODES;
        h1[i] = Hemb[(size_t)b * 4096 + j] + W1[(size_t)(1024 + c) * 4096 + j];
    }
}

// ---------------------------------------------------------------------------
// Attention logits: one wave per (node, head); lanes reduce over channels.
// ---------------------------------------------------------------------------
__global__ __launch_bounds__(256)
void att_logits_kernel(const float* __restrict__ h,
                       const float* __restrict__ a_src,
                       const float* __restrict__ a_dst,
                       float* __restrict__ al_s, float* __restrict__ al_d,
                       int H, int CH) {
    int wid = blockIdx.x * 8 + (threadIdx.x >> 5);
    if (wid >= NNODES * H) return;
    int lane = threadIdx.x & 31;
    int n = wid / H, hh = wid - n * H;
    const float* hp = h + ((size_t)n * H + hh) * CH;
    const float* as = a_src + (size_t)hh * CH;
    const float* ad = a_dst + (size_t)hh * CH;
    float s1 = 0.f, s2 = 0.f;
    for (int c = lane; c < CH; c += 32) {
        float v = hp[c];
        s1 += v * as[c];
        s2 += v * ad[c];
    }
    #pragma unroll
    for (int off = 16; off; off >>= 1) {
        s1 += __shfl_xor(s1, off, 32);
        s2 += __shfl_xor(s2, off, 32);
    }
    if (lane == 0) { al_s[wid] = s1; al_d[wid] = s2; }
}

// ---------------------------------------------------------------------------
// Edge softmax + aggregation. One wave per (node, head).
// ---------------------------------------------------------------------------
__global__ __launch_bounds__(256)
void gat_aggregate_kernel(const float* __restrict__ h,
                          const float* __restrict__ al_s,
                          const float* __restrict__ al_d,
                          const int* __restrict__ ei, int Eb,
                          const float* __restrict__ bias,
                          float* __restrict__ out, int H, int CH) {
    int wid = blockIdx.x * 8 + (threadIdx.x >> 5);
    if (wid >= NNODES * H) return;
    int lane = threadIdx.x & 31;
    int n = wid / H, hh = wid - n * H;
    int b = n / CNODES, c = n - b * CNODES;
    int nbase = b * CNODES;
    float ald = al_d[wid];

    // pass 1: running max (self loop guarantees >= 1 incoming edge)
    float eself = lrelu02(al_s[wid] + ald);
    float m = eself;
    for (int e = 0; e < Eb; ++e)
        if (ei[Eb + e] == c) {
            int sn = nbase + ei[e];
            m = fmaxf(m, lrelu02(al_s[sn * H + hh] + ald));
        }

    // pass 2: numerators (vector over channels) + denominator
    const int per = CH >> 5; // channels per lane (<= 16)
    float chacc[16];
    float w = __expf(eself - m);
    float denom = w;
    const float* hself = h + ((size_t)n * H + hh) * CH;
    for (int j = 0; j < per; ++j) chacc[j] = w * hself[lane + (j << 5)];
    for (int e = 0; e < Eb; ++e)
        if (ei[Eb + e] == c) {
            int sn = nbase + ei[e];
            float we = __expf(lrelu02(al_s[sn * H + hh] + ald) - m);
            denom += we;
            const float* hs = h + ((size_t)sn * H + hh) * CH;
            for (int j = 0; j < per; ++j) chacc[j] += we * hs[lane + (j << 5)];
        }

    float inv = 1.f / denom;
    float* op = out + ((size_t)n * H + hh) * CH;
    for (int j = 0; j < per; ++j) {
        int ch = lane + (j << 5);
        float v = chacc[j] * inv;
        if (bias) v += bias[hh * CH + ch];
        op[ch] = v;
    }
}

// ---------------------------------------------------------------------------
// ELU then cast to bf16 (next layer's WMMA A operand)
// ---------------------------------------------------------------------------
__global__ void elu_bf16_kernel(const float* __restrict__ in,
                                unsigned short* __restrict__ out, size_t n) {
    size_t i = (size_t)blockIdx.x * blockDim.x + threadIdx.x;
    size_t st = (size_t)gridDim.x * blockDim.x;
    for (; i < n; i += st) out[i] = f32_to_bf16_rne(elu1(in[i]));
}

// ---------------------------------------------------------------------------
// Layer-3 epilogue: mean over heads, + bias, ELU -> x3 (NNODES x 64)
// ---------------------------------------------------------------------------
__global__ void mean_heads_kernel(const float* __restrict__ agg,
                                  const float* __restrict__ bias,
                                  float* __restrict__ x3, int H, int CH) {
    size_t total = (size_t)NNODES * CH;
    size_t i = (size_t)blockIdx.x * blockDim.x + threadIdx.x;
    size_t st = (size_t)gridDim.x * blockDim.x;
    for (; i < total; i += st) {
        int n = (int)(i / CH);
        int ch = (int)(i - (size_t)n * CH);
        float s = 0.f;
        for (int hh = 0; hh < H; ++hh)
            s += agg[((size_t)n * H + hh) * CH + ch];
        x3[i] = elu1(s / (float)H + bias[ch]);
    }
}

// ---------------------------------------------------------------------------
// Head: per-image node mean -> 64->32 relu -> 32->15 sigmoid
// ---------------------------------------------------------------------------
__global__ __launch_bounds__(64)
void head_kernel(const float* __restrict__ x3,
                 const float* __restrict__ W1, const float* __restrict__ b1,
                 const float* __restrict__ W2, const float* __restrict__ b2,
                 float* __restrict__ out) {
    __shared__ float g[64];
    __shared__ float hid[32];
    int b = blockIdx.x, t = threadIdx.x;
    float s = 0.f;
    for (int c = 0; c < CNODES; ++c)
        s += x3[((size_t)b * CNODES + c) * 64 + t];
    g[t] = s * (1.f / (float)CNODES);
    __syncthreads();
    if (t < 32) {
        float a = b1[t];
        for (int k = 0; k < 64; ++k) a += g[k] * W1[k * 32 + t];
        hid[t] = fmaxf(a, 0.f);
    }
    __syncthreads();
    if (t < CNODES) {
        float a = b2[t];
        for (int k = 0; k < 32; ++k) a += hid[k] * W2[k * CNODES + t];
        out[b * CNODES + t] = 1.f / (1.f + __expf(-a));
    }
}

// ---------------------------------------------------------------------------
// Host-side launch
// ---------------------------------------------------------------------------
extern "C" void kernel_launch(void* const* d_in, const int* in_sizes, int n_in,
                              void* d_out, int out_size, void* d_ws, size_t ws_size,
                              hipStream_t stream) {
    (void)n_in; (void)out_size; (void)ws_size;
    const float* emb = (const float*)d_in[0];
    const int*   ei  = (const int*)d_in[1];
    const float* W1  = (const float*)d_in[2];
    const float* a1s = (const float*)d_in[3];
    const float* a1d = (const float*)d_in[4];
    const float* b1  = (const float*)d_in[5];
    const float* W2  = (const float*)d_in[6];
    const float* a2s = (const float*)d_in[7];
    const float* a2d = (const float*)d_in[8];
    const float* b2  = (const float*)d_in[9];
    const float* W3  = (const float*)d_in[10];
    const float* a3s = (const float*)d_in[11];
    const float* a3d = (const float*)d_in[12];
    const float* b3  = (const float*)d_in[13];
    const float* cW1 = (const float*)d_in[14];
    const float* cb1 = (const float*)d_in[15];
    const float* cW2 = (const float*)d_in[16];
    const float* cb2 = (const float*)d_in[17];
    float* out = (float*)d_out;
    const int Eb = in_sizes[1] / 2;

    // ---- bump allocator over workspace ----
    char* ws = (char*)d_ws;
    auto alloc = [&](size_t bytes) -> void* {
        void* p = (void*)ws;
        ws += (bytes + 255) & ~(size_t)255;
        return p;
    };
    unsigned short* emb_bf = (unsigned short*)alloc((size_t)BATCH * 1024 * 2);
    unsigned short* W1t_bf = (unsigned short*)alloc((size_t)4096 * 1024 * 2); // [4096][1024]
    unsigned short* W2t_bf = (unsigned short*)alloc((size_t)2048 * 4096 * 2); // [2048][4096]
    unsigned short* W3t_bf = (unsigned short*)alloc((size_t)256 * 2048 * 2);  // [256][2048]
    float* Hemb    = (float*)alloc((size_t)BATCH * 4096 * 4);
    float* h_big   = (float*)alloc((size_t)NNODES * 4096 * 4); // h1 / h2 / h3
    float* agg_big = (float*)alloc((size_t)NNODES * 4096 * 4); // agg1 / agg2 / agg3
    unsigned short* x_bf = (unsigned short*)alloc((size_t)NNODES * 4096 * 2);
    float* al_s = (float*)alloc((size_t)NNODES * 8 * 4);
    float* al_d = (float*)alloc((size_t)NNODES * 8 * 4);
    float* x3   = (float*)alloc((size_t)NNODES * 64 * 4);

    const int CB = 256;   // elementwise block
    const int CG = 2048;  // elementwise grid (grid-stride)

    // ---- precision converts (+ weight transposes for scatter-free staging) --
    cvt_bf16_kernel<<<CG, CB, 0, stream>>>(emb, emb_bf, (size_t)BATCH * 1024);
    cvt_transpose_bf16_kernel<<<dim3(4096 / 32, 1024 / 32), 256, 0, stream>>>(
        W1, W1t_bf, 1024, 4096);  // rows 0..1023 of W1
    cvt_transpose_bf16_kernel<<<dim3(2048 / 32, 4096 / 32), 256, 0, stream>>>(
        W2, W2t_bf, 4096, 2048);
    cvt_transpose_bf16_kernel<<<dim3(256 / 32, 2048 / 32), 256, 0, stream>>>(
        W3, W3t_bf, 2048, 256);

    // ---- layer 1 (GEMM collapsed to per-image: 512x1024x4096) ----
    gemm_bf16_wmma_kernel<<<dim3(4096 / TN, BATCH / TM), 256, 0, stream>>>(
        emb_bf, W1t_bf, Hemb, BATCH, 4096, 1024);
    build_h1_kernel<<<CG, CB, 0, stream>>>(Hemb, W1, h_big, (size_t)NNODES * 4096);
    {
        int H = 8, CH = 512;
        int blocks = (NNODES * H + 7) / 8;
        att_logits_kernel<<<blocks, 256, 0, stream>>>(h_big, a1s, a1d, al_s, al_d, H, CH);
        gat_aggregate_kernel<<<blocks, 256, 0, stream>>>(h_big, al_s, al_d, ei, Eb,
                                                         b1, agg_big, H, CH);
    }
    elu_bf16_kernel<<<CG, CB, 0, stream>>>(agg_big, x_bf, (size_t)NNODES * 4096);

    // ---- layer 2 (7680x4096x2048, the heavy GEMM) ----
    gemm_bf16_wmma_kernel<<<dim3(2048 / TN, NNODES / TM), 256, 0, stream>>>(
        x_bf, W2t_bf, h_big, NNODES, 2048, 4096);
    {
        int H = 8, CH = 256;
        int blocks = (NNODES * H + 7) / 8;
        att_logits_kernel<<<blocks, 256, 0, stream>>>(h_big, a2s, a2d, al_s, al_d, H, CH);
        gat_aggregate_kernel<<<blocks, 256, 0, stream>>>(h_big, al_s, al_d, ei, Eb,
                                                         b2, agg_big, H, CH);
    }
    elu_bf16_kernel<<<CG, CB, 0, stream>>>(agg_big, x_bf, (size_t)NNODES * 2048);

    // ---- layer 3 (7680x2048x256, mean over 4 heads) ----
    gemm_bf16_wmma_kernel<<<dim3(256 / TN, NNODES / TM), 256, 0, stream>>>(
        x_bf, W3t_bf, h_big, NNODES, 256, 2048);
    {
        int H = 4, CH = 64;
        int blocks = (NNODES * H + 7) / 8;
        att_logits_kernel<<<blocks, 256, 0, stream>>>(h_big, a3s, a3d, al_s, al_d, H, CH);
        gat_aggregate_kernel<<<blocks, 256, 0, stream>>>(h_big, al_s, al_d, ei, Eb,
                                                         nullptr, agg_big, H, CH);
        mean_heads_kernel<<<CG, CB, 0, stream>>>(agg_big, b3, x3, H, CH);
    }

    // ---- classifier head ----
    head_kernel<<<BATCH, 64, 0, stream>>>(x3, cW1, cb1, cW2, cb2, out);
}
